// SingleStageDetector_49984829391450
// MI455X (gfx1250) — matrix operand order repaired and checked by reading, not compiled, not run on hardware
//
#include <hip/hip_runtime.h>
#include <hip/hip_bf16.h>
#include <math.h>

#define ANUM 9
#define CNUM 20
#define BATCH 32
#define CIN 1280
#define HH 28
#define WWID 28
#define HWP (HH*WWID)       // 784
#define HID 128
#define OUTC 65             // 5*9+20
#define OUTP 80             // padded to 5 WMMA tiles
#define NBOX 64
#define MIDX 4096
#define PROP (ANUM*HWP)     // 7056
#define KSTEP 64
#define KROUNDS (CIN/KSTEP) // 20
#define PTILES (HWP/16)     // 49

typedef __attribute__((ext_vector_type(16))) __bf16 v16bf;
typedef __attribute__((ext_vector_type(8)))  float  v8f;

union FragU {
  uint4 q[2];
  v16bf v;
};

__device__ __forceinline__ unsigned short f2bf(float f) {
  unsigned int u = __builtin_bit_cast(unsigned int, f);
  u += 0x7FFFu + ((u >> 16) & 1u);   // round-to-nearest-even
  return (unsigned short)(u >> 16);
}

// A fragment: 16x32 bf16, row-major [M][ldk] source.
// lane L: row = caller-supplied, halves 0..7 = K kb..kb+7, halves 8..15 = +16
__device__ __forceinline__ v16bf load_frag_a(const unsigned short* base, int row,
                                             int k0, int ldk, int lane) {
  int kb = k0 + ((lane >> 4) << 3);
  const unsigned short* p = base + (size_t)row * ldk + kb;
  FragU f;
  f.q[0] = *(const uint4*)(p);
  f.q[1] = *(const uint4*)(p + 16);
  return f.v;
}

// B fragment: 32x16 bf16, source stored as [N][ldk] (K contiguous per column).
// lane L: col = n, halves 0..7 = K kb..kb+7 (kb = k0 + (L>=16 ? 8 : 0)), halves 8..15 = +16
__device__ __forceinline__ v16bf load_frag_b(const unsigned short* base, int col,
                                             int k0, int ldk, int lane) {
  int kb = k0 + ((lane >> 4) << 3);
  const unsigned short* p = base + (size_t)col * ldk + kb;
  FragU f;
  f.q[0] = *(const uint4*)(p);
  f.q[1] = *(const uint4*)(p + 16);
  return f.v;
}

// ---------------------------------------------------------------------------
// Kernel 1: convert W1 (128x1280) and W2 (65x128, pad to 80x128) to bf16
// ---------------------------------------------------------------------------
__global__ __launch_bounds__(256) void convert_weights_kernel(
    const float* __restrict__ W1, const float* __restrict__ W2,
    unsigned short* __restrict__ W1bf, unsigned short* __restrict__ W2bf) {
  int i = blockIdx.x * 256 + threadIdx.x;
  const int N1 = HID * CIN;    // 163840
  const int N2 = OUTP * HID;   // 10240
  if (i < N1) {
    W1bf[i] = f2bf(W1[i]);
  } else if (i < N1 + N2) {
    int j = i - N1;
    int r = j >> 7, c = j & 127;
    W2bf[j] = (r < OUTC) ? f2bf(W2[r * HID + c]) : (unsigned short)0;
  }
}

// ---------------------------------------------------------------------------
// Kernel 2: fused head. One block = (batch b, 16-pixel tile), 8 waves.
// GEMM1: [128 x 1280] x [1280 x 16] via bf16 WMMA, bias + leaky-relu,
// GEMM2: [80 x 128] x [128 x 16] via bf16 WMMA (h tile lives in LDS),
// writes raw out (B,65,784) fp32 to workspace.
// ---------------------------------------------------------------------------
__global__ __launch_bounds__(256) void fused_head_kernel(
    const float* __restrict__ features,
    const unsigned short* __restrict__ W1bf,
    const unsigned short* __restrict__ W2bf,
    const float* __restrict__ b1,
    const float* __restrict__ b2,
    float* __restrict__ outBuf) {
  __shared__ __align__(16) unsigned short sF[16][72];    // [pixel][K=64] (+8 pad)
  __shared__ __align__(16) unsigned short sH[16][136];   // [pixel][K2=128] (+8 pad)

  int blk = blockIdx.x;
  int b  = blk / PTILES;
  int pt = blk - b * PTILES;
  int p0 = pt * 16;
  int tid  = threadIdx.x;
  int wave = tid >> 5;
  int lane = tid & 31;

  const float* fbase = features + (size_t)b * CIN * HWP + p0;
  int rowA = (wave << 4) + (lane & 15);     // output-channel row for this lane

  // staging coordinates: thread loads float2 at (k, p..p+1) and (k+32, p..p+1)
  int sk = tid >> 3;            // 0..31
  int sp = (tid & 7) << 1;      // 0,2,...,14

  v8f acc = {0.f, 0.f, 0.f, 0.f, 0.f, 0.f, 0.f, 0.f};

  for (int kt = 0; kt < KROUNDS; ++kt) {
    int k0 = kt * KSTEP;
    __syncthreads();   // previous round's LDS reads complete
    // stage 64(k) x 16(pixel) feature tile -> sF[pixel][k], bf16
    {
      const float* src = fbase + (size_t)(k0 + sk) * HWP + sp;
      float2 f01 = *(const float2*)(src);
      float2 f23 = *(const float2*)(src + (size_t)32 * HWP);
      sF[sp    ][sk]      = f2bf(f01.x);
      sF[sp + 1][sk]      = f2bf(f01.y);
      sF[sp    ][sk + 32] = f2bf(f23.x);
      sF[sp + 1][sk + 32] = f2bf(f23.y);
    }
    // prefetch next round's tile (64 k-rows, one lane each); speculative
    if (kt + 1 < KROUNDS && tid < KSTEP) {
      __builtin_prefetch(fbase + (size_t)(k0 + KSTEP + tid) * HWP, 0, 3);
    }
    __syncthreads();
#pragma unroll
    for (int s = 0; s < 2; ++s) {
      v16bf af = load_frag_a(W1bf, rowA, k0 + s * 32, CIN, lane);
      v16bf bfr = load_frag_b(&sF[0][0], lane & 15, s * 32, 72, lane);
      acc = __builtin_amdgcn_wmma_f32_16x16x32_bf16(false, af, false, bfr,
                                                    (short)0, acc, false, false);
    }
  }

  // bias + leaky relu -> sH[pixel][o] bf16 (packed pair stores)
  {
    int p = lane & 15;
    int obase = (wave << 4) + ((lane >> 4) << 3);
#pragma unroll
    for (int v = 0; v < 8; v += 2) {
      int o = obase + v;
      float v0 = acc[v]     + b1[o];
      float v1 = acc[v + 1] + b1[o + 1];
      v0 = v0 > 0.f ? v0 : 0.01f * v0;
      v1 = v1 > 0.f ? v1 : 0.01f * v1;
      unsigned int pk = (unsigned int)f2bf(v0) | ((unsigned int)f2bf(v1) << 16);
      *(unsigned int*)&sH[p][o] = pk;
    }
  }
  __syncthreads();

  // GEMM2: waves 0..4 each compute a 16-row tile of the 80 (padded) outputs
  if (wave < 5) {
    v8f acc2 = {0.f, 0.f, 0.f, 0.f, 0.f, 0.f, 0.f, 0.f};
#pragma unroll
    for (int it = 0; it < HID / 32; ++it) {
      int k0 = it * 32;
      v16bf af = load_frag_a(W2bf, rowA, k0, HID, lane);
      v16bf bfr = load_frag_b(&sH[0][0], lane & 15, k0, 136, lane);
      acc2 = __builtin_amdgcn_wmma_f32_16x16x32_bf16(false, af, false, bfr,
                                                     (short)0, acc2, false, false);
    }
    int p = lane & 15;
    int obase = (wave << 4) + ((lane >> 4) << 3);
    float* optr = outBuf + (size_t)b * OUTC * HWP + p0 + p;
#pragma unroll
    for (int v = 0; v < 8; ++v) {
      int o = obase + v;
      if (o < OUTC) optr[(size_t)o * HWP] = acc2[v] + b2[o];
    }
  }
}

// ---------------------------------------------------------------------------
// Kernel 3: IoU matrix (B, 7056, 64) -> d_out[106496 ...]
// ---------------------------------------------------------------------------
__global__ __launch_bounds__(256) void iou_kernel(
    const float* __restrict__ bboxes,   // (B, 64, 5)
    const float* __restrict__ anc,      // (9, 2)
    float* __restrict__ iou_out) {
  __shared__ float sBB[NBOX * 5];
  __shared__ float sAnc[ANUM * 2];
  long long e0 = (long long)blockIdx.x * 256;
  int b = (int)(e0 / ((long long)PROP * NBOX));   // uniform per block
  int tid = threadIdx.x;
  for (int i = tid; i < NBOX * 5; i += 256) sBB[i] = bboxes[(size_t)b * NBOX * 5 + i];
  if (tid < ANUM * 2) sAnc[tid] = anc[tid];
  __syncthreads();

  long long e = e0 + tid;
  int n = (int)(e & 63);
  int p = (int)((e >> 6) % PROP);
  int a  = p / HWP;
  int hw = p - a * HWP;
  int y = hw / WWID, x = hw - y * WWID;
  float cx = x + 0.5f, cy = y + 0.5f;
  float hwd = sAnc[a * 2]     * 0.5f;
  float hht = sAnc[a * 2 + 1] * 0.5f;
  float px1 = cx - hwd, py1 = cy - hht, px2 = cx + hwd, py2 = cy + hht;
  const float* bb = &sBB[n * 5];
  float bx1 = bb[0], by1 = bb[1], bx2 = bb[2], by2 = bb[3];
  float s_p = (px2 - px1) * (py2 - py1);
  float s_b = (bx2 - bx1) * (by2 - by1);
  float ix1 = fmaxf(px1, bx1), iy1 = fmaxf(py1, by1);
  float ix2 = fminf(px2, bx2), iy2 = fminf(py2, by2);
  float s_i = fmaxf(ix2 - ix1, 0.f) * fmaxf(iy2 - iy1, 0.f);
  float s_u = s_p + s_b - s_i;
  float iou = fmaxf(s_i / (s_u + 1e-8f), 0.f);
  bool invalid = (s_u <= 0.f) || (s_p <= 0.f) || (s_b <= 0.f) || (bx1 < 0.f);
  iou_out[e] = invalid ? 0.f : iou;
}

// ---------------------------------------------------------------------------
// Kernel 4: gather + sigmoid transforms -> d_out[0 .. 106496)
//   [0,8192)       conf_scores (pos then neg), sigmoid
//   [8192,24576)   offsets (4096 x 4), xy: sigmoid-0.5, wh: raw
//   [24576,106496) class_scores (4096 x 20), raw
// ---------------------------------------------------------------------------
__global__ __launch_bounds__(256) void gather_kernel(
    const float* __restrict__ outBuf,
    const int* __restrict__ posIdx,
    const int* __restrict__ negIdx,
    float* __restrict__ dout) {
  int i = blockIdx.x * 256 + threadIdx.x;
  const int TOT = 8192 + MIDX * 4 + MIDX * CNUM;
  if (i >= TOT) return;
  float res;
  if (i < 8192) {
    int m = i & 4095;
    int id = (i < 4096) ? posIdx[m] : negIdx[m];
    int b = id / PROP, rem = id % PROP;
    int a = rem / HWP, hw = rem % HWP;
    float raw = outBuf[(size_t)b * OUTC * HWP + (size_t)(5 * a) * HWP + hw];
    res = 1.f / (1.f + expf(-raw));
  } else if (i < 24576) {
    int t = i - 8192;
    int m = t >> 2, d = t & 3;
    int id = posIdx[m];
    int b = id / PROP, rem = id % PROP;
    int a = rem / HWP, hw = rem % HWP;
    float raw = outBuf[(size_t)b * OUTC * HWP + (size_t)(5 * a + 1 + d) * HWP + hw];
    res = (d < 2) ? (1.f / (1.f + expf(-raw)) - 0.5f) : raw;
  } else {
    int t = i - 24576;
    int m = t / CNUM, c = t - m * CNUM;
    int id = posIdx[m];
    int b = id / PROP, rem = id % PROP;
    int hw = rem % HWP;
    res = outBuf[(size_t)b * OUTC * HWP + (size_t)(5 * ANUM + c) * HWP + hw];
  }
  dout[i] = res;
}

// ---------------------------------------------------------------------------
extern "C" void kernel_launch(void* const* d_in, const int* in_sizes, int n_in,
                              void* d_out, int out_size, void* d_ws, size_t ws_size,
                              hipStream_t stream) {
  const float* features = (const float*)d_in[0];
  // d_in[1] = grid (recomputed analytically, unused)
  const float* anc    = (const float*)d_in[2];
  const float* bboxes = (const float*)d_in[3];
  const int*   posIdx = (const int*)d_in[4];
  const int*   negIdx = (const int*)d_in[5];
  const float* W1 = (const float*)d_in[6];
  const float* b1 = (const float*)d_in[7];
  const float* W2 = (const float*)d_in[8];
  const float* b2 = (const float*)d_in[9];
  float* dout = (float*)d_out;

  // workspace layout
  unsigned short* W1bf = (unsigned short*)d_ws;                    // 163840 u16
  unsigned short* W2bf = W1bf + HID * CIN;                         // 10240 u16
  float* outBuf = (float*)((char*)d_ws + (size_t)(HID * CIN + OUTP * HID) * 2); // 348160 B, 16-aligned

  // 1. weight conversion (163840 + 10240 elems)
  convert_weights_kernel<<<(HID * CIN + OUTP * HID) / 256, 256, 0, stream>>>(
      W1, W2, W1bf, W2bf);

  // 2. fused GEMM1 + act + GEMM2  (32 batches x 49 pixel tiles)
  fused_head_kernel<<<BATCH * PTILES, 256, 0, stream>>>(
      features, W1bf, W2bf, b1, b2, outBuf);

  // 3. IoU matrix  (32*7056*64 = 14450688 elems, exactly 56448 blocks)
  iou_kernel<<<(BATCH * (long long)PROP * NBOX) / 256, 256, 0, stream>>>(
      bboxes, anc, dout + (8192 + MIDX * 4 + MIDX * CNUM));

  // 4. gathers + sigmoid transforms (106496 elems)
  gather_kernel<<<(8192 + MIDX * 4 + MIDX * CNUM + 255) / 256, 256, 0, stream>>>(
      outBuf, posIdx, negIdx, dout);
}
